// ClusterNormCholesky_37151467110663
// MI455X (gfx1250) — matrix operand order repaired and checked by reading, not compiled, not run on hardware
//
#include <hip/hip_runtime.h>
#include <hip/hip_bf16.h>

typedef __attribute__((ext_vector_type(16))) _Float16 v16h;
typedef __attribute__((ext_vector_type(8)))  float    v8f;
typedef __attribute__((ext_vector_type(2)))  _Float16 h2;
typedef int i4v __attribute__((vector_size(16)));      // matches builtin param type

constexpr int CDIM = 64;      // channels
constexpr int MDIM = 4096;    // samples per row
constexpr int MT   = 256;     // M-chunk processed per LDS stage
constexpr int LSX  = 264;     // xs row stride (halves) for Gram pass  [CDIM][LSX]
constexpr int LSXT = 72;      // xs row stride (halves) for Z pass     [MT][LSXT]
constexpr int CS   = 66;      // fp32 64x64 matrix row stride (floats)
constexpr int NT   = 256;     // threads per block (8 wave32)

#if __has_builtin(__builtin_amdgcn_global_load_async_to_lds_b128)
#define HAVE_ASYNC_LDS 1
#else
#define HAVE_ASYNC_LDS 0
#endif

__device__ __forceinline__ void wait_async0() {
#if __has_builtin(__builtin_amdgcn_s_wait_asynccnt)
  __builtin_amdgcn_s_wait_asynccnt(0);
#elif HAVE_ASYNC_LDS
  asm volatile("s_wait_asynccnt 0x0" ::: "memory");
#endif
}

// DMA one 64x256 fp32 chunk (64 KB) from global into LDS xraw.
// 4096 x 16B segments over 256 threads -> 16 async b128 ops per thread.
__device__ __forceinline__ void stage_chunk(const float* __restrict__ xg, int m0,
                                            float* xraw, int tid) {
#pragma unroll
  for (int i = 0; i < 16; ++i) {
    const int seg = i * NT + tid;          // 0..4095
    const int row = seg >> 6;              // 64 segments (of 4 floats) per row
    const int c4  = (seg & 63) << 2;
    const float* gp = xg + (size_t)row * MDIM + m0 + c4;
    float* lp = xraw + row * MT + c4;
#if HAVE_ASYNC_LDS
    __builtin_amdgcn_global_load_async_to_lds_b128(
        (__attribute__((address_space(1))) i4v*)gp,
        (__attribute__((address_space(3))) i4v*)lp, 0, 0);
#else
    *(float4*)lp = *(const float4*)gp;
#endif
  }
}

// ---- WMMA fragment loaders (gfx1250 wave32 layouts, ISA 7.12.2) ----

// A (16x32 f16): lanes 0-15 & 16-31 both hold rows row0..row0+15; lane-half
// selects K offset. VGPR v holds K pair: (v>>2)*16 + half*8 + (v&3)*2.
__device__ __forceinline__ v16h frag_a_f16(const _Float16* buf, int row0, int k0,
                                           int stride, int lane) {
  const int hlf = lane >> 4;
  const _Float16* rp = buf + (row0 + (lane & 15)) * stride;
  v16h a;
#pragma unroll
  for (int v = 0; v < 8; ++v) {
    int kk = k0 + ((v >> 2) << 4) + (hlf << 3) + ((v & 3) << 1);
    h2 p = *(const h2*)(rp + kk);
    a[2 * v] = p[0];
    a[2 * v + 1] = p[1];
  }
  return a;
}

// B (32x16 f16) where B(k,n) = buf[n][k]: column n = lane&15,
// K = (v>>2)*16 + half*8 + (v&3)*2 + j (pairs packed per VGPR).
__device__ __forceinline__ v16h frag_b_rows(const _Float16* buf, int n0, int k0,
                                            int stride, int lane) {
  const int hlf = lane >> 4;
  const _Float16* rp = buf + (n0 + (lane & 15)) * stride;
  v16h b;
#pragma unroll
  for (int v = 0; v < 8; ++v) {
    int kk = k0 + ((v >> 2) << 4) + (hlf << 3) + ((v & 3) << 1);
    h2 p = *(const h2*)(rp + kk);
    b[2 * v] = p[0];
    b[2 * v + 1] = p[1];
  }
  return b;
}

// A fragment of L^T: A(c,k) = L[k][c] for k>=c (L lower-tri fp32 in LDS), else 0.
__device__ __forceinline__ v16h frag_a_LT(const float* L, int c0, int k0, int lane) {
  const int hlf = lane >> 4;
  const int c = c0 + (lane & 15);
  v16h a;
#pragma unroll
  for (int v = 0; v < 8; ++v) {
    int kb = k0 + ((v >> 2) << 4) + (hlf << 3) + ((v & 3) << 1);
#pragma unroll
    for (int j = 0; j < 2; ++j) {
      int k = kb + j;
      a[2 * v + j] = (_Float16)((k >= c) ? L[k * CS + c] : 0.0f);
    }
  }
  return a;
}

// Parallel in-place Cholesky of SPD 64x64 (stride CS); lower triangle = factor.
__device__ __forceinline__ void cholesky64(float* A, int tid) {
  for (int k = 0; k < CDIM; ++k) {
    if (tid == 0) A[k * CS + k] = sqrtf(A[k * CS + k]);
    __syncthreads();
    float dk = A[k * CS + k];
    for (int i = k + 1 + tid; i < CDIM; i += NT) A[i * CS + k] /= dk;
    __syncthreads();
    for (int idx = tid; idx < CDIM * CDIM; idx += NT) {
      int i = idx >> 6, j = idx & 63;
      if (j > k && j <= i) A[i * CS + j] -= A[i * CS + k] * A[j * CS + k];
    }
    __syncthreads();
  }
}

__global__ __launch_bounds__(NT) void ClusterNormCholesky_kernel(
    const float* __restrict__ x, float* __restrict__ out) {
  __shared__ float    xraw[CDIM * MT];   // 64 KB fp32 async-DMA staging
  __shared__ _Float16 xs[MT * LSXT];     // 36864 B f16; aliased as [CDIM][LSX]
  __shared__ float m1[CDIM * CS];        // cov -> R -> Ainv -> L
  __shared__ float m2[CDIM * CS];        // Rinv
  __shared__ float s_mean[CDIM];
  __shared__ float s_scal[2];            // trace, tr(C@C)

  const int tid = threadIdx.x;
  const int lane = tid & 31;
  const int wave = tid >> 5;
  const int hlf = lane >> 4;
  const int b = blockIdx.x;
  const float* __restrict__ xg = x + (size_t)b * CDIM * MDIM;
  float* __restrict__ og = out + (size_t)b * CDIM * MDIM;

  if (tid < CDIM) s_mean[tid] = 0.0f;
  if (tid < 2) s_scal[tid] = 0.0f;

  // prologue DMA for chunk 0 (overlaps nothing yet, but primes the pipeline)
  stage_chunk(xg, 0, xraw, tid);
  __syncthreads();

  // ---------------- Pass 1: Gram matrix G = X X^T via WMMA + row sums -------
  // 16 output tiles (4x4 of 16x16); wave w owns tiles 2w and 2w+1 (same row tile).
  const int ct = wave >> 1;
  const int dt0 = (wave & 1) * 2;
  v8f acc0 = {};
  v8f acc1 = {};

  for (int m0 = 0; m0 < MDIM; m0 += MT) {
    wait_async0();
    __syncthreads();                       // xraw holds chunk m0
    // convert fp32 -> f16 stage + per-row partial sums
#pragma unroll 8
    for (int r = 0; r < CDIM; ++r) {
      float v = xraw[r * MT + tid];
      xs[r * LSX + tid] = (_Float16)v;
#pragma unroll
      for (int off = 16; off; off >>= 1) v += __shfl_xor(v, off, 32);
      if (lane == 0) atomicAdd(&s_mean[r], v);
    }
    __syncthreads();                       // xraw drained -> safe to refill
    if (m0 + MT < MDIM) stage_chunk(xg, m0 + MT, xraw, tid);  // overlap w/ WMMA
#pragma unroll
    for (int ks = 0; ks < MT / 32; ++ks) {
      const int k0 = ks * 32;
      v16h a = frag_a_f16(xs, ct * 16, k0, LSX, lane);
      v16h b0 = frag_b_rows(xs, dt0 * 16, k0, LSX, lane);
      v16h b1 = frag_b_rows(xs, (dt0 + 1) * 16, k0, LSX, lane);
      acc0 = __builtin_amdgcn_wmma_f32_16x16x32_f16(false, a, false, b0,
                                                    (short)0, acc0, false, false);
      acc1 = __builtin_amdgcn_wmma_f32_16x16x32_f16(false, a, false, b1,
                                                    (short)0, acc1, false, false);
    }
    __syncthreads();                       // xs reads done before next convert
  }

  // prefetch pass-2 chunk 0 under the solver's shadow (xraw is free now)
  stage_chunk(xg, 0, xraw, tid);

  if (tid < CDIM) s_mean[tid] *= (1.0f / MDIM);
  __syncthreads();

  // cov = G/M - mu mu^T  (C/D layout: VGPR r -> row r + half*8, col = lane&15)
  {
    const float invM = 1.0f / MDIM;
    const int col0 = dt0 * 16 + (lane & 15);
    const int col1 = (dt0 + 1) * 16 + (lane & 15);
#pragma unroll
    for (int r = 0; r < 8; ++r) {
      int row = ct * 16 + r + hlf * 8;
      m1[row * CS + col0] = acc0[r] * invM - s_mean[row] * s_mean[col0];
      m1[row * CS + col1] = acc1[r] * invM - s_mean[row] * s_mean[col1];
    }
  }
  __syncthreads();

  // ---------------- Rao-Blackwell Ledoit-Wolf shrinkage ---------------------
  if (tid < CDIM) atomicAdd(&s_scal[0], m1[tid * CS + tid]);
  {
    float st = 0.0f;
    for (int idx = tid; idx < CDIM * CDIM; idx += NT) {
      int i = idx >> 6, j = idx & 63;
      st += m1[i * CS + j] * m1[j * CS + i];
    }
    atomicAdd(&s_scal[1], st);
  }
  __syncthreads();
  {
    const float tr = s_scal[0], st = s_scal[1];
    const float p = (float)CDIM, nn = (float)MDIM;
    const float num = (nn - 2.0f) / nn * st + tr * tr;
    const float den = (nn + 2.0f) * (st - tr * tr / p);
    const float rho = fminf(num / den, 1.0f);
    const float mu = tr / p;
    for (int idx = tid; idx < CDIM * CDIM; idx += NT) {
      int i = idx >> 6, j = idx & 63;
      float v = (1.0f - rho) * m1[i * CS + j];
      if (i == j) v += rho * mu;
      m1[i * CS + j] = v;
    }
  }
  __syncthreads();

  // ---------------- L = chol(inv(cov)) --------------------------------------
  cholesky64(m1, tid);                       // cov = R R^T, R lower in m1

  if (tid < CDIM) {                          // Rinv columns, 1 thread each
    const int j = tid;
    for (int i = j; i < CDIM; ++i) {
      float s = (i == j) ? 1.0f : 0.0f;
      for (int k = j; k < i; ++k) s -= m1[i * CS + k] * m2[k * CS + j];
      m2[i * CS + j] = s / m1[i * CS + i];
    }
  }
  __syncthreads();

  for (int idx = tid; idx < CDIM * CDIM; idx += NT) {  // Ainv = Rinv^T Rinv
    int i = idx >> 6, j = idx & 63;
    int kk = (i > j) ? i : j;
    float s = 0.0f;
    for (int k = kk; k < CDIM; ++k) s += m2[k * CS + i] * m2[k * CS + j];
    m1[i * CS + j] = s;
  }
  __syncthreads();

  cholesky64(m1, tid);                       // m1 lower = L, inv(cov) = L L^T

  // ---------------- Pass 2: Z = L^T @ (x - mu) via WMMA ---------------------
  v16h la[4][2];
#pragma unroll
  for (int c = 0; c < 4; ++c)
#pragma unroll
    for (int ks = 0; ks < 2; ++ks) la[c][ks] = frag_a_LT(m1, c * 16, ks * 32, lane);

  const int mt0 = wave * 2;                  // 16 m-tiles per chunk, 2 per wave
  for (int m0 = 0; m0 < MDIM; m0 += MT) {
    wait_async0();
    __syncthreads();                         // xraw holds chunk m0
#pragma unroll 8
    for (int r = 0; r < CDIM; ++r) {         // center + transpose into f16 stage
      float v = xraw[r * MT + tid] - s_mean[r];
      xs[tid * LSXT + r] = (_Float16)v;
    }
    __syncthreads();                         // xraw drained -> refill
    if (m0 + MT < MDIM) stage_chunk(xg, m0 + MT, xraw, tid);
#pragma unroll
    for (int t = 0; t < 2; ++t) {
      const int mt = mt0 + t;
      v16h b0 = frag_b_rows(xs, mt * 16, 0, LSXT, lane);
      v16h b1 = frag_b_rows(xs, mt * 16, 32, LSXT, lane);
      const int col = m0 + mt * 16 + (lane & 15);
#pragma unroll
      for (int c = 0; c < 4; ++c) {
        v8f z = {};
        z = __builtin_amdgcn_wmma_f32_16x16x32_f16(false, la[c][0], false, b0,
                                                   (short)0, z, false, false);
        z = __builtin_amdgcn_wmma_f32_16x16x32_f16(false, la[c][1], false, b1,
                                                   (short)0, z, false, false);
#pragma unroll
        for (int r = 0; r < 8; ++r) {
          int row = c * 16 + r + hlf * 8;
          og[(size_t)row * MDIM + col] = z[r];
        }
      }
    }
    __syncthreads();                         // xs reads done before next convert
  }
}

extern "C" void kernel_launch(void* const* d_in, const int* in_sizes, int n_in,
                              void* d_out, int out_size, void* d_ws, size_t ws_size,
                              hipStream_t stream) {
  (void)n_in; (void)out_size; (void)d_ws; (void)ws_size;
  const float* x = (const float*)d_in[0];
  float* out = (float*)d_out;
  const int B = in_sizes[0] / (CDIM * MDIM);   // 256 for the reference shapes
  ClusterNormCholesky_kernel<<<dim3(B), dim3(NT), 0, stream>>>(x, out);
}
